// SequenceModel_27960237097299
// MI455X (gfx1250) — matrix operand order repaired and checked by reading, not compiled, not run on hardware
//
#include <hip/hip_runtime.h>
#include <hip/hip_bf16.h>

// ---------------------------------------------------------------------------
// 2-layer LSTM (B=256, T=512, F=32, H=1024) + projection on MI455X (gfx1250).
// f16 WMMA with fp32 accumulation; weights L2-resident (~25 MB f16 vs 192 MB).
// Recurrence is launch/latency bound -> fuse pipeline-parallel stages
// {layer2(t), layer1(t+1), proj(t-1)} into one kernel per time step.
// Each wave owns a 32(M)x16(U) tile across all 4 gates: A-frags reused 4x,
// B(weight)-frags reused 2x -> 1.5 b128 loads per wmma.
// ---------------------------------------------------------------------------

typedef __attribute__((ext_vector_type(16))) _Float16 v16h;
typedef __attribute__((ext_vector_type(8)))  _Float16 v8h;
typedef __attribute__((ext_vector_type(8)))  float    v8f;

#define BQ 256
#define TQ 512
#define FQ 32
#define HQ 1024

#define CELL_BLOCKS 64   // 8 m-groups (32 rows) x 8 u-blocks
#define PROJ_BLOCKS 4

// ---- WMMA fragment loaders (CDNA5 ISA 7.12.2 VGPR layouts) -----------------
// A-matrix 16x32 f16, M x K row-major, lda in elements.
// lane l<16 : m=l,    VGPR0..3 = K 0..7,  VGPR4..7 = K 16..23
// lane l>=16: m=l-16, VGPR0..3 = K 8..15, VGPR4..7 = K 24..31
__device__ __forceinline__ v16h load_fragA(const _Float16* __restrict__ A,
                                           int lda, int m0, int k0) {
  const int l   = threadIdx.x & 31;
  const int m   = m0 + (l & 15);
  const int klo = (l >> 4) << 3;               // 0 or 8
  const _Float16* p = A + (size_t)m * lda + k0 + klo;
  v8h lo = *(const v8h*)(p);                   // K: k0+klo+0..7
  v8h hi = *(const v8h*)(p + 16);              // K: k0+klo+16..23
  v16h r;
#pragma unroll
  for (int i = 0; i < 8; ++i) { r[i] = lo[i]; r[i + 8] = hi[i]; }
  return r;
}

// B-matrix 32x16 f16 from row-major W[N][K] (B = W^T), ldw in elements.
// lane l<16: n=l, K=k0+0..7,16..23; lane>=16: n=l-16, K=k0+8..15,24..31
__device__ __forceinline__ v16h load_fragB(const _Float16* __restrict__ W,
                                           int ldw, int n0, int k0) {
  const int l  = threadIdx.x & 31;
  const int n  = n0 + (l & 15);
  const int kb = k0 + ((l >> 4) << 3);         // 0 or 8
  const _Float16* p = W + (size_t)n * ldw + kb;
  v8h lo = *(const v8h*)(p);
  v8h hi = *(const v8h*)(p + 16);
  v16h r;
#pragma unroll
  for (int i = 0; i < 8; ++i) { r[i] = lo[i]; r[i + 8] = hi[i]; }
  return r;
}

__device__ __forceinline__ float fsigmoid(float x) {
  return 1.0f / (1.0f + __expf(-x));
}
__device__ __forceinline__ float ftanh(float x) {
  return 1.0f - 2.0f / (__expf(2.0f * x) + 1.0f);
}

// ---- Fused LSTM cell block -------------------------------------------------
// gates[b,*] = A1[b,:]@W1^T + A2[b,:]@W2^T + bias; nonlinearity + c/h update
// fused in-register. rb in [0,64): rb>>3 -> m-group (32 rows), rb&7 + wave ->
// 16-unit column tile. Per wave: acc[2 mtiles][4 gates].
__device__ __forceinline__ void cell_block(
    int rb,
    const _Float16* __restrict__ A1, int lda1, const _Float16* __restrict__ W1, int K1,
    const _Float16* __restrict__ A2, int lda2, const _Float16* __restrict__ W2, int K2,
    const float* __restrict__ bias, float* __restrict__ c,
    _Float16* __restrict__ hOut) {
  const int wave = threadIdx.x >> 5;
  const int m0   = (rb >> 3) << 5;                       // 32-row group
  const int u0   = (((rb & 7) << 3) + wave) << 4;        // 16-unit column

  v8f acc[2][4];
#pragma unroll
  for (int mi = 0; mi < 2; ++mi)
#pragma unroll
    for (int g = 0; g < 4; ++g) acc[mi][g] = {};

  for (int k0 = 0; k0 < K1; k0 += 32) {
    v16h a0 = load_fragA(A1, lda1, m0, k0);
    v16h a1 = load_fragA(A1, lda1, m0 + 16, k0);
#pragma unroll
    for (int g = 0; g < 4; ++g) {
      v16h b = load_fragB(W1, K1, g * HQ + u0, k0);
      acc[0][g] = __builtin_amdgcn_wmma_f32_16x16x32_f16(false, a0, false, b, (short)0, acc[0][g], false, false);
      acc[1][g] = __builtin_amdgcn_wmma_f32_16x16x32_f16(false, a1, false, b, (short)0, acc[1][g], false, false);
    }
  }
  for (int k0 = 0; k0 < K2; k0 += 32) {
    v16h a0 = load_fragA(A2, lda2, m0, k0);
    v16h a1 = load_fragA(A2, lda2, m0 + 16, k0);
#pragma unroll
    for (int g = 0; g < 4; ++g) {
      v16h b = load_fragB(W2, K2, g * HQ + u0, k0);
      acc[0][g] = __builtin_amdgcn_wmma_f32_16x16x32_f16(false, a0, false, b, (short)0, acc[0][g], false, false);
      acc[1][g] = __builtin_amdgcn_wmma_f32_16x16x32_f16(false, a1, false, b, (short)0, acc[1][g], false, false);
    }
  }

  // Epilogue: acc[mi][g][v] @ lane l -> row = m0+16*mi+v+8*(l>=16), col = u0+(l&15)
  const int   l   = threadIdx.x & 31;
  const int   col = u0 + (l & 15);
  const float b_i = bias[0 * HQ + col];
  const float b_f = bias[1 * HQ + col];
  const float b_g = bias[2 * HQ + col];
  const float b_o = bias[3 * HQ + col];
#pragma unroll
  for (int mi = 0; mi < 2; ++mi) {
#pragma unroll
    for (int v = 0; v < 8; ++v) {
      const int    row = m0 + (mi << 4) + v + ((l >> 4) << 3);
      const size_t idx = (size_t)row * HQ + col;
      const float iv = fsigmoid(acc[mi][0][v] + b_i);
      const float fv = fsigmoid(acc[mi][1][v] + b_f);
      const float gv = ftanh(acc[mi][2][v] + b_g);
      const float ov = fsigmoid(acc[mi][3][v] + b_o);
      const float cn = fv * c[idx] + iv * gv;
      c[idx]    = cn;
      hOut[idx] = (_Float16)(ov * ftanh(cn));
    }
  }
}

// ---- Projection block: out[:,t,:] = h2 @ W_out^T + b_out -------------------
// M=256, N=32, K=1024 -> 32 tiles of 16x16 over 4 blocks x 8 waves.
__device__ __forceinline__ void proj_block(
    int rb, const _Float16* __restrict__ h2, const _Float16* __restrict__ Wout,
    const float* __restrict__ bout, float* __restrict__ out, int t) {
  const int wave = threadIdx.x >> 5;
  const int tile = (rb << 3) + wave;           // 0..31
  const int m0   = (tile >> 1) << 4;
  const int n0   = (tile & 1) << 4;

  v8f acc = {};
  for (int k0 = 0; k0 < HQ; k0 += 32) {
    v16h a = load_fragA(h2, HQ, m0, k0);
    v16h b = load_fragB(Wout, HQ, n0, k0);
    acc = __builtin_amdgcn_wmma_f32_16x16x32_f16(false, a, false, b, (short)0, acc, false, false);
  }
  const int   l   = threadIdx.x & 31;
  const int   col = n0 + (l & 15);
  const float bb  = bout[col];
#pragma unroll
  for (int v = 0; v < 8; ++v) {
    const int row = m0 + v + ((l >> 4) << 3);
    out[(size_t)row * (TQ * FQ) + (size_t)t * FQ + col] = acc[v] + bb;
  }
}

// ---- Kernels ---------------------------------------------------------------
__global__ void __launch_bounds__(256) lstm_cell_kernel(
    const _Float16* __restrict__ A1, int lda1, const _Float16* __restrict__ W1, int K1,
    const _Float16* __restrict__ A2, int lda2, const _Float16* __restrict__ W2, int K2,
    const float* __restrict__ bias, float* __restrict__ c,
    _Float16* __restrict__ hOut) {
  cell_block(blockIdx.x, A1, lda1, W1, K1, A2, lda2, W2, K2, bias, c, hOut);
}

__global__ void __launch_bounds__(256) proj_kernel(
    const _Float16* __restrict__ h2, const _Float16* __restrict__ Wout,
    const float* __restrict__ bout, float* __restrict__ out, int t) {
  proj_block(blockIdx.x, h2, Wout, bout, out, t);
}

// One kernel per time step: blocks [0,64) run layer2(t); [64,64+nB) run
// layer1(t+1) (independent of layer2(t)); remaining blocks run proj(t-1).
// Role is uniform per block -> EXEC stays all-ones for every WMMA.
__global__ void __launch_bounds__(256) fused_step_kernel(
    // cell A: layer2(t)
    const _Float16* __restrict__ A1a, int lda1a, const _Float16* __restrict__ W1a, int K1a,
    const _Float16* __restrict__ A2a, int lda2a, const _Float16* __restrict__ W2a, int K2a,
    const float* __restrict__ biasA, float* __restrict__ cA, _Float16* __restrict__ hOutA,
    // cell B: layer1(t+1)
    int nB,
    const _Float16* __restrict__ A1b, int lda1b, const _Float16* __restrict__ W1b, int K1b,
    const _Float16* __restrict__ A2b, int lda2b, const _Float16* __restrict__ W2b, int K2b,
    const float* __restrict__ biasB, float* __restrict__ cB, _Float16* __restrict__ hOutB,
    // proj: step t-1
    const _Float16* __restrict__ h2prev, const _Float16* __restrict__ Wout,
    const float* __restrict__ bout, float* __restrict__ out, int tprev) {
  const int b = blockIdx.x;
  if (b < CELL_BLOCKS) {
    cell_block(b, A1a, lda1a, W1a, K1a, A2a, lda2a, W2a, K2a, biasA, cA, hOutA);
  } else if (b < CELL_BLOCKS + nB) {
    cell_block(b - CELL_BLOCKS, A1b, lda1b, W1b, K1b, A2b, lda2b, W2b, K2b, biasB, cB, hOutB);
  } else {
    proj_block(b - CELL_BLOCKS - nB, h2prev, Wout, bout, out, tprev);
  }
}

// ---- Setup helpers ---------------------------------------------------------
__global__ void f32_to_f16_kernel(const float* __restrict__ src,
                                  _Float16* __restrict__ dst, int n) {
  for (int i = blockIdx.x * blockDim.x + threadIdx.x; i < n;
       i += gridDim.x * blockDim.x)
    dst[i] = (_Float16)src[i];
}

__global__ void bias_sum_kernel(const float* __restrict__ a,
                                const float* __restrict__ b,
                                float* __restrict__ o, int n) {
  for (int i = blockIdx.x * blockDim.x + threadIdx.x; i < n;
       i += gridDim.x * blockDim.x)
    o[i] = a[i] + b[i];
}

// ---------------------------------------------------------------------------
extern "C" void kernel_launch(void* const* d_in, const int* in_sizes, int n_in,
                              void* d_out, int out_size, void* d_ws, size_t ws_size,
                              hipStream_t stream) {
  (void)in_sizes; (void)n_in; (void)out_size; (void)ws_size;

  const float* x      = (const float*)d_in[0];
  const float* W_ih1  = (const float*)d_in[1];
  const float* W_hh1  = (const float*)d_in[2];
  const float* b_ih1  = (const float*)d_in[3];
  const float* b_hh1  = (const float*)d_in[4];
  const float* W_ih2  = (const float*)d_in[5];
  const float* W_hh2  = (const float*)d_in[6];
  const float* b_ih2  = (const float*)d_in[7];
  const float* b_hh2  = (const float*)d_in[8];
  const float* W_out  = (const float*)d_in[9];
  const float* b_out  = (const float*)d_in[10];
  float*       out    = (float*)d_out;

  // Workspace carve-up (256B aligned; total ~38.2 MB)
  char*  ws  = (char*)d_ws;
  size_t off = 0;
  auto alloc = [&](size_t bytes) -> void* {
    void* p = ws + off;
    off += (bytes + 255) & ~(size_t)255;
    return p;
  };
  _Float16* xf16 = (_Float16*)alloc((size_t)BQ * TQ * FQ * 2);
  _Float16* wih1 = (_Float16*)alloc((size_t)4 * HQ * FQ * 2);
  _Float16* whh1 = (_Float16*)alloc((size_t)4 * HQ * HQ * 2);
  _Float16* wih2 = (_Float16*)alloc((size_t)4 * HQ * HQ * 2);
  _Float16* whh2 = (_Float16*)alloc((size_t)4 * HQ * HQ * 2);
  _Float16* wout = (_Float16*)alloc((size_t)FQ * HQ * 2);
  float*    b1   = (float*)alloc((size_t)4 * HQ * 4);
  float*    b2   = (float*)alloc((size_t)4 * HQ * 4);
  _Float16* h1buf[2] = {(_Float16*)alloc((size_t)BQ * HQ * 2),
                        (_Float16*)alloc((size_t)BQ * HQ * 2)};
  _Float16* h2buf[2] = {(_Float16*)alloc((size_t)BQ * HQ * 2),
                        (_Float16*)alloc((size_t)BQ * HQ * 2)};
  float*    c1   = (float*)alloc((size_t)BQ * HQ * 4);
  float*    c2   = (float*)alloc((size_t)BQ * HQ * 4);

  // One-time per-call setup: f16 copies, fused biases, zero state.
  f32_to_f16_kernel<<<1024, 256, 0, stream>>>(x,     xf16, BQ * TQ * FQ);
  f32_to_f16_kernel<<<128,  256, 0, stream>>>(W_ih1, wih1, 4 * HQ * FQ);
  f32_to_f16_kernel<<<1024, 256, 0, stream>>>(W_hh1, whh1, 4 * HQ * HQ);
  f32_to_f16_kernel<<<1024, 256, 0, stream>>>(W_ih2, wih2, 4 * HQ * HQ);
  f32_to_f16_kernel<<<1024, 256, 0, stream>>>(W_hh2, whh2, 4 * HQ * HQ);
  f32_to_f16_kernel<<<128,  256, 0, stream>>>(W_out, wout, FQ * HQ);
  bias_sum_kernel<<<16, 256, 0, stream>>>(b_ih1, b_hh1, b1, 4 * HQ);
  bias_sum_kernel<<<16, 256, 0, stream>>>(b_ih2, b_hh2, b2, 4 * HQ);
  hipMemsetAsync(h1buf[0], 0, (size_t)BQ * HQ * 2, stream);
  hipMemsetAsync(h2buf[0], 0, (size_t)BQ * HQ * 2, stream);
  hipMemsetAsync(c1, 0, (size_t)BQ * HQ * 4, stream);
  hipMemsetAsync(c2, 0, (size_t)BQ * HQ * 4, stream);

  // Buffer convention: h1(t) lives in h1buf[(t+1)&1]; h2(t) in h2buf[(t+1)&1].
  // Prologue: layer1(0) -> h1(0)
  lstm_cell_kernel<<<CELL_BLOCKS, 256, 0, stream>>>(
      xf16, TQ * FQ, wih1, FQ,
      h1buf[0], HQ, whh1, HQ, b1, c1, h1buf[1]);

  for (int t = 0; t < TQ; ++t) {
    _Float16* h1_t   = h1buf[(t + 1) & 1];  // h1(t)   (read: layer2(t), layer1(t+1))
    _Float16* h1_tp1 = h1buf[t & 1];        // h1(t+1) (write)
    _Float16* h2_tm1 = h2buf[t & 1];        // h2(t-1) (read: layer2(t), proj(t-1))
    _Float16* h2_t   = h2buf[(t + 1) & 1];  // h2(t)   (write)

    const int nB    = (t + 1 < TQ) ? CELL_BLOCKS : 0;   // layer1(t+1) exists?
    const int nProj = (t >= 1) ? PROJ_BLOCKS : 0;       // proj(t-1) exists?

    fused_step_kernel<<<CELL_BLOCKS + nB + nProj, 256, 0, stream>>>(
        // layer2(t): gates = h1(t)@W_ih2^T + h2(t-1)@W_hh2^T + b2
        h1_t, HQ, wih2, HQ, h2_tm1, HQ, whh2, HQ, b2, c2, h2_t,
        // layer1(t+1): gates = x_{t+1}@W_ih1^T + h1(t)@W_hh1^T + b1
        nB,
        xf16 + (size_t)(t + 1) * FQ, TQ * FQ, wih1, FQ,
        h1_t, HQ, whh1, HQ, b1, c1, h1_tp1,
        // proj(t-1): out[:,t-1,:] = h2(t-1)@W_out^T + b_out
        h2_tm1, wout, b_out, out, t - 1);
  }

  // Epilogue: proj(T-1)
  proj_kernel<<<PROJ_BLOCKS, 256, 0, stream>>>(
      h2buf[TQ & 1], wout, b_out, out, TQ - 1);
}
